// GE2ELoss_1219770712687
// MI455X (gfx1250) — compile-verified, matrix operand order checked by reading, not run on hardware
//
#include <hip/hip_runtime.h>
#include <hip/hip_bf16.h>
#include <math.h>

#define N_SPK 1024
#define M_UTT 10
#define D_DIM 768
#define GEPS  1e-6f

typedef __attribute__((ext_vector_type(16))) __bf16 v16bf;
typedef __attribute__((ext_vector_type(8)))  __bf16 v8bf;
typedef __attribute__((ext_vector_type(8)))  float  v8f;

union BfFrag { v16bf v; v8bf h[2]; };

__global__ void ge2e_zero(float* out) { out[0] = 0.0f; }

__device__ __forceinline__ void wave_reduce_into(float x, float* dst) {
  #pragma unroll
  for (int off = 16; off > 0; off >>= 1) x += __shfl_xor(x, off, 32);
  if ((threadIdx.x & 31) == 0) atomicAdd(dst, x);
}

// Kernel 1: one pass over dvecs. Per speaker n (one block):
//   sums over M, centroid, centroid norm, per-utterance norm,
//   leave-one-out centroid norm and dot.
// Emits: Abf  = dvec rows normalized to unit length (bf16, 10240 x 768)
//        Cbf  = centroids normalized to unit length  (bf16, 1024 x 768)
//        own  = w * cos(dvec, excl-centroid) + b     (f32, 10240)
__global__ void __launch_bounds__(256) ge2e_prep(
    const float* __restrict__ dvecs, const float* __restrict__ wp,
    const float* __restrict__ bp, __bf16* __restrict__ Abf,
    __bf16* __restrict__ Cbf, float* __restrict__ own) {
  __shared__ float red[31];
  const int n   = blockIdx.x;
  const int tid = threadIdx.x;
  if (tid < 31) red[tid] = 0.0f;
  __syncthreads();

  const float* base = dvecs + (size_t)n * M_UTT * D_DIM;
  float v[3][M_UTT];
  float s[3], cv[3];
  #pragma unroll
  for (int i = 0; i < 3; ++i) {
    const int d = tid + i * 256;
    float a = 0.0f;
    #pragma unroll
    for (int m = 0; m < M_UTT; ++m) { float x = base[m * D_DIM + d]; v[i][m] = x; a += x; }
    s[i] = a;
  }
  float c2p = 0.0f;
  #pragma unroll
  for (int i = 0; i < 3; ++i) { cv[i] = s[i] * (1.0f / M_UTT); c2p += cv[i] * cv[i]; }

  float dn2[M_UTT], en2[M_UTT], de[M_UTT];
  #pragma unroll
  for (int m = 0; m < M_UTT; ++m) {
    float a = 0.0f, b = 0.0f, c = 0.0f;
    #pragma unroll
    for (int i = 0; i < 3; ++i) {
      float x = v[i][m];
      float e = (s[i] - x) * (1.0f / (M_UTT - 1));
      a += x * x; b += e * e; c += x * e;
    }
    dn2[m] = a; en2[m] = b; de[m] = c;
  }

  wave_reduce_into(c2p, &red[0]);
  #pragma unroll
  for (int m = 0; m < M_UTT; ++m) wave_reduce_into(dn2[m], &red[1 + m]);
  #pragma unroll
  for (int m = 0; m < M_UTT; ++m) wave_reduce_into(en2[m], &red[11 + m]);
  #pragma unroll
  for (int m = 0; m < M_UTT; ++m) wave_reduce_into(de[m], &red[21 + m]);
  __syncthreads();

  const float invcn = 1.0f / fmaxf(sqrtf(red[0]), GEPS);
  #pragma unroll
  for (int i = 0; i < 3; ++i) {
    int d = tid + i * 256;
    Cbf[(size_t)n * D_DIM + d] = (__bf16)(cv[i] * invcn);
  }
  #pragma unroll
  for (int m = 0; m < M_UTT; ++m) {
    float invdn = 1.0f / fmaxf(sqrtf(red[1 + m]), GEPS);
    #pragma unroll
    for (int i = 0; i < 3; ++i) {
      int d = tid + i * 256;
      Abf[((size_t)(n * M_UTT + m)) * D_DIM + d] = (__bf16)(v[i][m] * invdn);
    }
  }
  if (tid < M_UTT) {
    const int m = tid;
    float dn = fmaxf(sqrtf(red[1 + m]), GEPS);
    float en = fmaxf(sqrtf(red[11 + m]), GEPS);
    float so = red[21 + m] / (dn * en);
    own[n * M_UTT + m] = wp[0] * so + bp[0];
  }
}

// Kernel 2: fused sim-matmul + diagonal substitution + logsumexp + loss sum.
// Block = 16 A-rows (staged in LDS) x all 1024 centroids.
// 8 waves; wave w covers column tiles [8w, 8w+8). K is the outer loop so
// bf16 WMMA fragments are re-loaded each iteration (bounded registers).
__global__ void __launch_bounds__(256) ge2e_loss(
    const __bf16* __restrict__ Abf, const __bf16* __restrict__ Cbf,
    const float* __restrict__ own, const float* __restrict__ wp,
    const float* __restrict__ bp, float* __restrict__ out) {
  __shared__ __bf16 Ash[16 * D_DIM];      // 24 KB of 320 KB WGP LDS
  __shared__ float Msh[8][16];
  __shared__ float Ssh[8][16];

  const int tid     = threadIdx.x;
  const int wave    = tid >> 5;
  const int lane    = tid & 31;
  const int half    = lane >> 4;
  const int nloc    = lane & 15;
  const int rowbase = blockIdx.x * 16;

  // Stage this block's 16 normalized utterance rows into LDS (16B chunks).
  {
    const v8bf* gsrc = (const v8bf*)(Abf + (size_t)rowbase * D_DIM);
    v8bf* ldst = (v8bf*)Ash;
    #pragma unroll
    for (int i = 0; i < 6; ++i) ldst[tid + i * 256] = gsrc[tid + i * 256];
  }
  __syncthreads();

  v8f acc[8] = {};

  // 16-bit A fragment layout: lanes 0-15 hold K 0..7 & 16..23; lanes 16-31
  // hold K 8..15 & 24..31 (same pattern for B columns).
  const __bf16* arow = Ash + nloc * D_DIM + (half ? 8 : 0);

  for (int kc = 0; kc < 24; ++kc) {          // K = 24 * 32 = 768
    BfFrag a;
    a.h[0] = *(const v8bf*)(arow + kc * 32);
    a.h[1] = *(const v8bf*)(arow + kc * 32 + 16);
    #pragma unroll
    for (int t = 0; t < 8; ++t) {
      const int col = (wave * 8 + t) * 16 + nloc;
      const __bf16* bp2 = Cbf + (size_t)col * D_DIM + (half ? 8 : 0) + kc * 32;
      if (kc + 1 < 24) __builtin_prefetch((const void*)(bp2 + 32), 0, 3);
      BfFrag b;
      b.h[0] = *(const v8bf*)(bp2);
      b.h[1] = *(const v8bf*)(bp2 + 16);
      acc[t] = __builtin_amdgcn_wmma_f32_16x16x32_bf16(
          false, a.v, false, b.v, (short)0, acc[t], false, false);
    }
  }

  // Epilogue: C layout is VGPR j -> row (j + 8*half), lane -> column nloc.
  const float wv = wp[0], bv = bp[0];
  #pragma unroll
  for (int j = 0; j < 8; ++j) {
    const int rg   = rowbase + j + 8 * half;   // global utterance row
    const int nrow = rg / M_UTT;               // its speaker id (diag column)
    float vals[8];
    float m = -__builtin_inff();
    #pragma unroll
    for (int t = 0; t < 8; ++t) {
      const int kcol = (wave * 8 + t) * 16 + nloc;
      float vv = acc[t][j] * wv + bv;
      if (kcol == nrow) vv = own[rg];          // diagonal substitution
      vals[t] = vv;
      m = fmaxf(m, vv);
    }
    #pragma unroll
    for (int off = 1; off < 16; off <<= 1) m = fmaxf(m, __shfl_xor(m, off, 16));
    float sv = 0.0f;
    #pragma unroll
    for (int t = 0; t < 8; ++t) sv += __expf(vals[t] - m);
    #pragma unroll
    for (int off = 1; off < 16; off <<= 1) sv += __shfl_xor(sv, off, 16);
    if (nloc == 0) { Msh[wave][j + 8 * half] = m; Ssh[wave][j + 8 * half] = sv; }
  }
  __syncthreads();

  if (tid < 16) {
    float m = -__builtin_inff(), sm = 0.0f;
    #pragma unroll
    for (int w2 = 0; w2 < 8; ++w2) {
      float om = Msh[w2][tid], os = Ssh[w2][tid];
      float nm = fmaxf(m, om);
      sm = sm * __expf(m - nm) + os * __expf(om - nm);
      m = nm;
    }
    // -log_softmax at own column = logsumexp - own value
    float rl = m + __logf(sm) - own[rowbase + tid];
    atomicAdd(out, rl);
  }
}

extern "C" void kernel_launch(void* const* d_in, const int* in_sizes, int n_in,
                              void* d_out, int out_size, void* d_ws, size_t ws_size,
                              hipStream_t stream) {
  const float* dvecs = (const float*)d_in[0];
  const float* w     = (const float*)d_in[1];
  const float* b     = (const float*)d_in[2];
  float* out = (float*)d_out;

  char* ws = (char*)d_ws;
  const size_t abytes = (size_t)N_SPK * M_UTT * D_DIM * sizeof(__bf16); // 15.7 MB
  const size_t cbytes = (size_t)N_SPK * D_DIM * sizeof(__bf16);         // 1.5 MB
  __bf16* Abf = (__bf16*)ws;
  __bf16* Cbf = (__bf16*)(ws + abytes);
  float*  own = (float*)(ws + abytes + cbytes);

  ge2e_zero<<<1, 1, 0, stream>>>(out);
  ge2e_prep<<<N_SPK, 256, 0, stream>>>(dvecs, w, b, Abf, Cbf, own);
  ge2e_loss<<<(N_SPK * M_UTT) / 16, 256, 0, stream>>>(Abf, Cbf, own, w, b, out);
}
